// PerceiverAttention_87505663688925
// MI455X (gfx1250) — matrix operand order repaired and checked by reading, not compiled, not run on hardware
//
#include <hip/hip_runtime.h>
#include <hip/hip_bf16.h>

// ---------------------------------------------------------------------------
// PerceiverAttention for MI455X (gfx1250, wave32, WMMA).
// fp32 in -> LN fp32 -> bf16 GEMMs (v_wmma_f32_16x16x32_bf16, f32 acc)
// -> fp32 softmax (folding the post-softmax /sqrt(DH)) -> fp32 out.
// Projections use pre-transposed bf16 weights so B tiles are staged into LDS
// with the CDNA5 async global->LDS DMA path (ASYNCcnt) with zero VALU work.
// ---------------------------------------------------------------------------

typedef __bf16 bf16_t;
typedef __attribute__((ext_vector_type(16))) __bf16        v16bf;
typedef __attribute__((ext_vector_type(8)))  __bf16        v8bf;
typedef __attribute__((ext_vector_type(8)))  float         v8f;
typedef __attribute__((ext_vector_type(8)))  unsigned short v8us;

#define BB   8
#define TT   8
#define BT   64          // B*T
#define NN_  1024        // media tokens
#define NL_  64          // latents
#define NK   1088        // N + NL
#define DIM_ 1024
#define HH   16
#define DH_  64
#define HID_ 1024
#define KVC  2048        // 2*HID

// ---- helpers ---------------------------------------------------------------

__device__ __forceinline__ unsigned short f2bf(float f) {
  union { float f; unsigned int u; } c; c.f = f;
  unsigned int u = c.u;
  u += 0x7FFFu + ((u >> 16) & 1u);       // round-to-nearest-even
  return (unsigned short)(u >> 16);
}

// A fragment: 16x32 bf16 row-major tile (contiguous k order, matching the B fragments).
// lane<16: row=lane,   K = 0..15
// lane>=16: row=lane-16, K = 16..31
__device__ __forceinline__ v16bf load_a_frag(const unsigned short* A, int lda) {
  int lane = threadIdx.x & 31;
  int r    = lane & 15;
  int kb   = (lane >> 4) << 4;           // 0 or 16
  const unsigned short* p = A + r * lda + kb;
  v16bf f;
  *(v8bf*)&f       = *(const v8bf*)(p);       // K = kb .. kb+7
  *((v8bf*)&f + 1) = *(const v8bf*)(p + 8);   // K = kb+8 .. kb+15
  return f;
}

// B fragment from LDS tile stored [n][k] (n in [0,64), k in [0,32)).
// lane: col n = (lane&15)+n0, K = (lane>>4)*16 + j  (contiguous -> 32B read)
__device__ __forceinline__ v16bf load_b_frag_lds(const unsigned short* lds, int n0) {
  int lane = threadIdx.x & 31;
  int n    = (lane & 15) + n0;
  int kb   = (lane >> 4) << 4;           // 0 or 16
  return *(const v16bf*)(lds + n * 32 + kb);
}

// B fragment for NT GEMM (B supplied as [N,K] row-major): contiguous in K.
__device__ __forceinline__ v16bf load_b_frag_nt(const unsigned short* Bt, int ldb) {
  int lane = threadIdx.x & 31;
  int n    = lane & 15;
  int kb   = (lane >> 4) << 4;
  const unsigned short* p = Bt + n * ldb + kb;
  v16bf f;
  *(v8bf*)&f       = *(const v8bf*)(p);
  *((v8bf*)&f + 1) = *(const v8bf*)(p + 8);
  return f;
}

// C/D fragment store: lane col n = lane&15, rows = v + 8*(lane>=16)
__device__ __forceinline__ void store_c(float* C, int ldc, v8f a) {
  int lane = threadIdx.x & 31;
  int n    = lane & 15;
  int mb   = (lane >> 4) << 3;
#pragma unroll
  for (int i = 0; i < 8; ++i) C[(mb + i) * ldc + n] = a[i];
}
__device__ __forceinline__ void store_c(unsigned short* C, int ldc, v8f a) {
  int lane = threadIdx.x & 31;
  int n    = lane & 15;
  int mb   = (lane >> 4) << 3;
#pragma unroll
  for (int i = 0; i < 8; ++i) C[(mb + i) * ldc + n] = f2bf(a[i]);
}

__device__ __forceinline__ v8f wmma_bf16(v16bf a, v16bf b, v8f c) {
  return __builtin_amdgcn_wmma_f32_16x16x32_bf16(
      /*neg_a=*/false, a, /*neg_b=*/false, b,
      /*c_mod=*/(short)0, c, /*reuse_a=*/false, /*reuse_b=*/false);
}

// Cooperative VALU stage of a 32(K) x 64(N) bf16 B-tile into LDS, transposed
// to lds[n*32+k] (used where B can't be pre-transposed, i.e. V tiles).
__device__ __forceinline__ void stage_b_tile(const unsigned short* B, int ldb,
                                             unsigned short* lds) {
  for (int e = threadIdx.x; e < 256; e += blockDim.x) {
    int k  = e >> 3;                 // 0..31
    int n8 = (e & 7) << 3;           // 0,8,..,56
    v8us row = *(const v8us*)(B + k * ldb + n8);   // coalesced 16B global load
#pragma unroll
    for (int j = 0; j < 8; ++j) lds[(n8 + j) * 32 + k] = row[j];
  }
}

// ---- elementwise / LN / softmax kernels ------------------------------------

// bf16-convert AND transpose a weight: src [K,N] f32 -> dst [N,K] bf16.
// 16x16 tile per 256-thread block; reads coalesced, writes tiny (weights only).
__global__ void cvt_transpose_bf16(const float* __restrict__ src,
                                   unsigned short* __restrict__ dst,
                                   int K, int N) {
  int n = blockIdx.x * 16 + (threadIdx.x & 15);
  int k = blockIdx.y * 16 + (threadIdx.x >> 4);
  dst[(size_t)n * K + k] = f2bf(src[(size_t)k * N + n]);
}

// One 256-thread block per row of 1024. Writes bf16 into the concatenated
// kv-input buffer X[bt*1088 + dst_off + i].
__global__ void layernorm_bf16(const float* __restrict__ X,
                               const float* __restrict__ g,
                               const float* __restrict__ b,
                               unsigned short* __restrict__ Y,
                               int rows_per_bt, int dst_stride, int dst_off) {
  int r  = blockIdx.x;
  int bt = r / rows_per_bt;
  int i  = r % rows_per_bt;
  const float* x = X + (size_t)r * DIM_;
  unsigned short* y = Y + (size_t)(bt * dst_stride + dst_off + i) * DIM_;

  int t = threadIdx.x;
  float4 v = *(const float4*)(x + t * 4);
  float s  = v.x + v.y + v.z + v.w;
  float s2 = v.x * v.x + v.y * v.y + v.z * v.z + v.w * v.w;
#pragma unroll
  for (int off = 16; off; off >>= 1) {
    s  += __shfl_xor(s,  off, 32);
    s2 += __shfl_xor(s2, off, 32);
  }
  __shared__ float rs[8], rs2[8];
  if ((t & 31) == 0) { rs[t >> 5] = s; rs2[t >> 5] = s2; }
  __syncthreads();
  float ts = 0.f, ts2 = 0.f;
#pragma unroll
  for (int j = 0; j < 8; ++j) { ts += rs[j]; ts2 += rs2[j]; }
  float mean = ts * (1.0f / DIM_);
  float var  = ts2 * (1.0f / DIM_) - mean * mean;
  float rstd = rsqrtf(var + 1e-5f);
  const float* xv = &v.x;
#pragma unroll
  for (int j = 0; j < 4; ++j) {
    int c = t * 4 + j;
    y[c] = f2bf((xv[j] - mean) * rstd * g[c] + b[c]);
  }
}

// Wave-per-row softmax over 1088 with the post-softmax /sqrt(DH) folded in.
__global__ void softmax_scale(const float* __restrict__ S,
                              unsigned short* __restrict__ P, int nrows) {
  int wid  = (int)((blockIdx.x * blockDim.x + threadIdx.x) >> 5);
  int lane = threadIdx.x & 31;
  if (wid >= nrows) return;
  const float* s = S + (size_t)wid * NK;
  unsigned short* p = P + (size_t)wid * NK;

  float x[34];
  float m = -3.0e38f;
#pragma unroll
  for (int i = 0; i < 34; ++i) { x[i] = s[lane + i * 32]; m = fmaxf(m, x[i]); }
#pragma unroll
  for (int off = 16; off; off >>= 1) m = fmaxf(m, __shfl_xor(m, off, 32));
  float sum = 0.f;
#pragma unroll
  for (int i = 0; i < 34; ++i) { x[i] = __expf(x[i] - m); sum += x[i]; }
#pragma unroll
  for (int off = 16; off; off >>= 1) sum += __shfl_xor(sum, off, 32);
  float inv = 1.0f / (sum * 8.0f);               // SCALE = sqrt(64) = 8
#pragma unroll
  for (int i = 0; i < 34; ++i) p[lane + i * 32] = f2bf(x[i] * inv);
}

// ---- GEMM kernels ----------------------------------------------------------

// C[M,N] = A[M,K](bf16) @ Bt[N,K](bf16)^T  with Bt row-major [N,K].
// Block = 256 thr = 8 waves; block tile 256x64; wave tile 32x64 (2 A frags,
// 8 WMMAs per K-step). The 64x32 B tile is ALREADY [n][k]-contiguous in
// memory, so it is staged into LDS with the CDNA5 async global->LDS DMA
// (one global_load_async_to_lds_b128 per thread per K-step, ASYNCcnt).
// A rows may be group-strided: phys_row = abase + (r/rpg)*gstride + (r%rpg).
// Requires M%256==0, N%64==0, K%32==0 (true for every call here).
template <typename CT>
__global__ void gemm_nt(const unsigned short* __restrict__ A, int lda,
                        int rpg, int gstride, int abase,
                        const unsigned short* __restrict__ Bt,
                        CT* __restrict__ C, int ldc, int K) {
  __shared__ __align__(32) unsigned short ldsB[64 * 32];
  int w  = threadIdx.x >> 5;
  int m0 = blockIdx.y * 256 + w * 32;
  int n0 = blockIdx.x * 64;
  int prow = abase + (m0 / rpg) * gstride + (m0 % rpg);
  const unsigned short* Aw = A + (size_t)prow * lda;

  // async-stage addressing: thread t copies 16B = Bt[n0+sn][k0+sk .. +7]
  int t  = threadIdx.x;
  int sn = t >> 2;                 // 0..63
  int sk = (t & 3) << 3;           // 0,8,16,24
  const unsigned short* gsrc = Bt + (size_t)(n0 + sn) * K + sk;
  // generic->LDS: low 32 bits of a generic LDS pointer are the LDS byte offset
  unsigned lds_dst = (unsigned)(uintptr_t)ldsB + (unsigned)(sn * 32 + sk) * 2u;

  v8f z = {0, 0, 0, 0, 0, 0, 0, 0};
  v8f acc[8] = {z, z, z, z, z, z, z, z};

  for (int k0 = 0; k0 < K; k0 += 32) {
    unsigned long long ga = (unsigned long long)(uintptr_t)(gsrc + k0);
    asm volatile("global_load_async_to_lds_b128 %0, %1, off"
                 :: "v"(lds_dst), "v"(ga) : "memory");
    asm volatile("s_wait_asynccnt 0x0" ::: "memory");
    __syncthreads();                               // tile visible to all waves
    if (k0 + 64 < K) __builtin_prefetch(Aw + k0 + 64, 0, 0);
    v16bf a0 = load_a_frag(Aw + k0, lda);
    v16bf a1 = load_a_frag(Aw + 16 * lda + k0, lda);
#pragma unroll
    for (int q = 0; q < 4; ++q) {
      v16bf b = load_b_frag_lds(ldsB, q * 16);
      acc[q]     = wmma_bf16(a0, b, acc[q]);
      acc[4 + q] = wmma_bf16(a1, b, acc[4 + q]);
    }
    __syncthreads();                               // done reading before next DMA
  }
  CT* Cw = C + (size_t)m0 * ldc + n0;
#pragma unroll
  for (int q = 0; q < 4; ++q) {
    store_c(Cw + q * 16,            ldc, acc[q]);
    store_c(Cw + 16 * ldc + q * 16, ldc, acc[4 + q]);
  }
}

// S[z, 64, 1088] = Q_h[64,64] @ K_h[1088,64]^T   (z = bt*H + h)
// NT form: per-lane B fragments contiguous in K -> direct global b128 loads.
// Block = 128 thr = 4 waves (16 q-rows each); grid = (1088/64, BT*H).
__global__ void attn_score(const unsigned short* __restrict__ Q,
                           const unsigned short* __restrict__ KV,
                           float* __restrict__ S) {
  int z  = blockIdx.y;
  int bt = z >> 4;
  int h  = z & 15;
  const unsigned short* Aq = Q  + (size_t)bt * NL_ * HID_ + h * DH_;
  const unsigned short* Bk = KV + (size_t)bt * NK * KVC   + h * DH_;
  float* Sz = S + (size_t)z * NL_ * NK;

  int w  = threadIdx.x >> 5;
  int m0 = w * 16;
  int n0 = blockIdx.x * 64;

  v8f zz = {0, 0, 0, 0, 0, 0, 0, 0};
  v8f acc[4] = {zz, zz, zz, zz};
#pragma unroll
  for (int k0 = 0; k0 < DH_; k0 += 32) {
    v16bf a = load_a_frag(Aq + m0 * HID_ + k0, HID_);
#pragma unroll
    for (int q = 0; q < 4; ++q) {
      v16bf b = load_b_frag_nt(Bk + (size_t)(n0 + q * 16) * KVC + k0, KVC);
      acc[q] = wmma_bf16(a, b, acc[q]);
    }
  }
#pragma unroll
  for (int q = 0; q < 4; ++q)
    store_c(Sz + (size_t)m0 * NK + n0 + q * 16, NK, acc[q]);
}

// O_h[64,64] = P[64,1088] @ V_h[1088,64]. Block = 128 thr = 4 waves; grid = BT*H.
// V is an activation (can't pre-transpose for free) -> VALU transpose staging.
__global__ void attn_pv(const unsigned short* __restrict__ P,
                        const unsigned short* __restrict__ KV,
                        unsigned short* __restrict__ O) {
  int z  = blockIdx.x;
  int bt = z >> 4;
  int h  = z & 15;
  const unsigned short* Pz = P  + (size_t)z * NL_ * NK;
  const unsigned short* V  = KV + (size_t)bt * NK * KVC + HID_ + h * DH_;
  unsigned short* Oz = O + (size_t)bt * NL_ * HID_ + h * DH_;

  __shared__ __align__(32) unsigned short ldsBt[64 * 32];
  int w  = threadIdx.x >> 5;
  int m0 = w * 16;

  v8f zz = {0, 0, 0, 0, 0, 0, 0, 0};
  v8f acc[4] = {zz, zz, zz, zz};
  for (int k0 = 0; k0 < NK; k0 += 32) {
    __syncthreads();
    stage_b_tile(V + (size_t)k0 * KVC, KVC, ldsBt);
    __syncthreads();
    v16bf a = load_a_frag(Pz + m0 * NK + k0, NK);
#pragma unroll
    for (int q = 0; q < 4; ++q)
      acc[q] = wmma_bf16(a, load_b_frag_lds(ldsBt, q * 16), acc[q]);
  }
#pragma unroll
  for (int q = 0; q < 4; ++q)
    store_c(Oz + (size_t)m0 * HID_ + q * 16, HID_, acc[q]);
}

// ---- host launcher ---------------------------------------------------------

extern "C" void kernel_launch(void* const* d_in, const int* in_sizes, int n_in,
                              void* d_out, int out_size, void* d_ws, size_t ws_size,
                              hipStream_t stream) {
  const float* media   = (const float*)d_in[0];
  const float* latents = (const float*)d_in[1];
  const float* g_m     = (const float*)d_in[2];
  const float* b_m     = (const float*)d_in[3];
  const float* g_l     = (const float*)d_in[4];
  const float* b_l     = (const float*)d_in[5];
  const float* Wq      = (const float*)d_in[6];
  const float* Wkv     = (const float*)d_in[7];
  const float* Wout    = (const float*)d_in[8];
  float* out = (float*)d_out;

  // workspace carve-out (~880 MB total)
  char* ws = (char*)d_ws;
  size_t off = 0;
  auto alloc = [&](size_t bytes) -> void* {
    void* p = ws + off; off += (bytes + 255) & ~(size_t)255; return p;
  };
  unsigned short* WqT_b   = (unsigned short*)alloc((size_t)DIM_ * HID_ * 2); // [HID,DIM]
  unsigned short* WkvT_b  = (unsigned short*)alloc((size_t)DIM_ * KVC  * 2); // [KVC,DIM]
  unsigned short* WoutT_b = (unsigned short*)alloc((size_t)HID_ * DIM_ * 2); // [DIM,HID]
  unsigned short* Xb      = (unsigned short*)alloc((size_t)BT * NK  * DIM_ * 2);
  unsigned short* Qb      = (unsigned short*)alloc((size_t)BT * NL_ * HID_ * 2);
  unsigned short* KVb     = (unsigned short*)alloc((size_t)BT * NK  * KVC  * 2);
  float*          Sb      = (float*)         alloc((size_t)BT * HH * NL_ * NK * 4);
  unsigned short* Pb      = (unsigned short*)alloc((size_t)BT * HH * NL_ * NK * 2);
  unsigned short* Ob      = (unsigned short*)alloc((size_t)BT * NL_ * HID_ * 2);

  // 1. weights -> bf16, transposed to [N,K] for NT GEMMs
  cvt_transpose_bf16<<<dim3(HID_/16, DIM_/16), 256, 0, stream>>>(Wq,   WqT_b,   DIM_, HID_);
  cvt_transpose_bf16<<<dim3(KVC /16, DIM_/16), 256, 0, stream>>>(Wkv,  WkvT_b,  DIM_, KVC);
  cvt_transpose_bf16<<<dim3(DIM_/16, HID_/16), 256, 0, stream>>>(Wout, WoutT_b, HID_, DIM_);

  // 2. LayerNorm -> concatenated bf16 X[bt, 0:1024]=media_ln, X[bt, 1024:1088]=latents_ln
  layernorm_bf16<<<BT * NN_, 256, 0, stream>>>(media,   g_m, b_m, Xb, NN_, NK, 0);
  layernorm_bf16<<<BT * NL_, 256, 0, stream>>>(latents, g_l, b_l, Xb, NL_, NK, NN_);

  // 3. KV = X @ Wkv (bf16);  Q = X_lat @ Wq (bf16)   [async-DMA staged NT GEMM]
  gemm_nt<unsigned short><<<dim3(KVC/64, (BT*NK)/256), 256, 0, stream>>>(
      Xb, DIM_, 1 << 30, 0, 0, WkvT_b, KVb, KVC, DIM_);
  gemm_nt<unsigned short><<<dim3(HID_/64, (BT*NL_)/256), 256, 0, stream>>>(
      Xb, DIM_, NL_, NK, NN_, WqT_b, Qb, HID_, DIM_);

  // 4. scores S = Q_h K_h^T  (f32)
  attn_score<<<dim3(NK/64, BT*HH), 128, 0, stream>>>(Qb, KVb, Sb);

  // 5. softmax / SCALE -> bf16 P
  softmax_scale<<<(BT*HH*NL_*32 + 255)/256, 256, 0, stream>>>(Sb, Pb, BT*HH*NL_);

  // 6. O_h = P V_h  (bf16)
  attn_pv<<<BT*HH, 128, 0, stream>>>(Pb, KVb, Ob);

  // 7. out = O @ Wout  (f32 -> d_out)
  gemm_nt<float><<<dim3(DIM_/64, (BT*NL_)/256), 256, 0, stream>>>(
      Ob, HID_, 1 << 30, 0, 0, WoutT_b, out, DIM_, HID_);
}
